// BertEncoderLayer_35622458753888
// MI455X (gfx1250) — compile-verified, matrix operand order checked by reading, not run on hardware
//
#include <hip/hip_runtime.h>
#include <hip/hip_bf16.h>
#include <cstdint>
#include <cstddef>

#define H_DIM 768
#define NHEAD 12
#define HD 64
#define FF_DIM 3072
#define SEQ 2048
#define BATCH 4
#define MTOT (BATCH * SEQ) /* 8192 */
#define LN_EPS 1e-12f

typedef __attribute__((ext_vector_type(16))) _Float16 v16h;
typedef __attribute__((ext_vector_type(8)))  _Float16 v8h;
typedef __attribute__((ext_vector_type(8)))  float    v8f;

__device__ __forceinline__ v8f vzero8() {
    v8f z;
#pragma unroll
    for (int i = 0; i < 8; ++i) z[i] = 0.0f;
    return z;
}

__device__ __forceinline__ v8f wmma16(v16h a, v16h b, v8f c) {
    return __builtin_amdgcn_wmma_f32_16x16x32_f16(false, a, false, b, (short)0, c, false, false);
}

// ds_swizzle xor-shuffle within 32 lanes (masks <=8 stay inside 16-lane halves)
template <int MASK>
__device__ __forceinline__ float swz_xor(float x) {
    int i = __builtin_bit_cast(int, x);
    i = __builtin_amdgcn_ds_swizzle(i, (MASK << 10) | 0x1f);
    return __builtin_bit_cast(float, i);
}

__device__ __forceinline__ void rowmax8(float (&a)[8]) {
#pragma unroll
    for (int r = 0; r < 8; ++r) {
        a[r] = fmaxf(a[r], swz_xor<1>(a[r]));
        a[r] = fmaxf(a[r], swz_xor<2>(a[r]));
        a[r] = fmaxf(a[r], swz_xor<4>(a[r]));
        a[r] = fmaxf(a[r], swz_xor<8>(a[r]));
    }
}

__device__ __forceinline__ void rowsum8(float (&a)[8]) {
#pragma unroll
    for (int r = 0; r < 8; ++r) {
        a[r] += swz_xor<1>(a[r]);
        a[r] += swz_xor<2>(a[r]);
        a[r] += swz_xor<4>(a[r]);
        a[r] += swz_xor<8>(a[r]);
    }
}

// Load a 16x32 f16 fragment (A-layout; B fragments use the same layout from an
// N-major tile). lane l: row = row0 + (l&15); elements 0..7 <- K = kb+kh..,
// elements 8..15 <- K = kb+16+kh.., kh = (l>>4)*8.
__device__ __forceinline__ v16h ld_frag(const _Float16* base, int row0, int ldm, int kb) {
    int lane = threadIdx.x & 31;
    const _Float16* p = base + (size_t)(row0 + (lane & 15)) * ldm + kb + ((lane >> 4) << 3);
    v8h lo = *(const v8h*)(p);
    v8h hi = *(const v8h*)(p + 16);
    v16h r;
#pragma unroll
    for (int i = 0; i < 8; ++i) { r[i] = lo[i]; r[i + 8] = hi[i]; }
    return r;
}

// ---------------------------------------------------------------- conversion
__global__ void cvt_f32_f16(const float* __restrict__ in, _Float16* __restrict__ out, size_t n) {
    size_t i = (size_t)blockIdx.x * 1024 + threadIdx.x;
#pragma unroll
    for (int j = 0; j < 4; ++j, i += 256)
        if (i < n) out[i] = (_Float16)in[i];
}

// ---------------------------------------------------------------- GEMM
// C[M,N] = A[M,K] @ B[K,N] (+bias, epilogue). 128x128 tile, BK=32, 256 thr.
#define LDS_STRIDE 40  // padded halves per LDS row (80B, 16B aligned)

struct StageRegs { v8h a0, a1, b0, b1; };

__device__ __forceinline__ void gload_tiles(const _Float16* __restrict__ A,
                                            const _Float16* __restrict__ B,
                                            int bM, int bN, int K, int N, int kt, int nk,
                                            int tid, StageRegs& s) {
    int r = tid >> 2, c = (tid & 3) * 8;
    s.a0 = *(const v8h*)(A + (size_t)(bM + r) * K + kt * 32 + c);
    s.a1 = *(const v8h*)(A + (size_t)(bM + r + 64) * K + kt * 32 + c);
    int k = tid >> 4, nc = (tid & 15) * 8;
    s.b0 = *(const v8h*)(B + (size_t)(kt * 32 + k) * N + bN + nc);
    s.b1 = *(const v8h*)(B + (size_t)(kt * 32 + k + 16) * N + bN + nc);
    // warm L2 for the tile after next (speculative prefetch, OOB-safe)
    if (kt + 1 < nk) {
        __builtin_prefetch(A + (size_t)(bM + r) * K + (kt + 1) * 32 + c, 0, 1);
        __builtin_prefetch(B + (size_t)((kt + 1) * 32 + k) * N + bN + nc, 0, 1);
    }
}

__device__ __forceinline__ void store_tiles(_Float16* sA, _Float16* sB, int tid,
                                            const StageRegs& s) {
    int r = tid >> 2, c = (tid & 3) * 8;
    *(v8h*)(sA + r * LDS_STRIDE + c) = s.a0;
    *(v8h*)(sA + (r + 64) * LDS_STRIDE + c) = s.a1;
    int k = tid >> 4, nc = (tid & 15) * 8;
#pragma unroll
    for (int j = 0; j < 8; ++j) sB[(nc + j) * LDS_STRIDE + k] = s.b0[j];
#pragma unroll
    for (int j = 0; j < 8; ++j) sB[(nc + j) * LDS_STRIDE + k + 16] = s.b1[j];
}

// EPI: 0 = bias -> f16 out; 1 = bias + f32 residual -> f32 out; 2 = bias + exact GELU -> f16 out
template <int EPI>
__global__ __launch_bounds__(256, 1) void gemm_f16(
        const _Float16* __restrict__ A, const _Float16* __restrict__ B,
        const float* __restrict__ bias, const float* __restrict__ res,
        float* __restrict__ Cf, _Float16* __restrict__ Ch,
        int M, int N, int K) {
    __shared__ __align__(16) _Float16 sA[2][128 * LDS_STRIDE];
    __shared__ __align__(16) _Float16 sB[2][128 * LDS_STRIDE];

    const int tid = threadIdx.x, lane = tid & 31, wid = tid >> 5;
    const int wm = wid & 1, wn = wid >> 1;            // 2 x 4 wave grid
    const int bM = blockIdx.y * 128, bN = blockIdx.x * 128;

    v8f acc[4][2];
#pragma unroll
    for (int m = 0; m < 4; ++m)
#pragma unroll
        for (int n = 0; n < 2; ++n) acc[m][n] = vzero8();

    const int nk = K / 32;
    StageRegs sr;
    gload_tiles(A, B, bM, bN, K, N, 0, nk, tid, sr);
    store_tiles(sA[0], sB[0], tid, sr);
    __syncthreads();

    for (int kt = 0; kt < nk; ++kt) {
        const int buf = kt & 1;
        const bool pre = (kt + 1 < nk);
        if (pre) gload_tiles(A, B, bM, bN, K, N, kt + 1, nk, tid, sr);  // overlap with WMMA

        v16h af[4], bf[2];
#pragma unroll
        for (int m = 0; m < 4; ++m) af[m] = ld_frag(sA[buf], wm * 64 + m * 16, LDS_STRIDE, 0);
#pragma unroll
        for (int n = 0; n < 2; ++n) bf[n] = ld_frag(sB[buf], wn * 32 + n * 16, LDS_STRIDE, 0);
#pragma unroll
        for (int m = 0; m < 4; ++m)
#pragma unroll
            for (int n = 0; n < 2; ++n) acc[m][n] = wmma16(af[m], bf[n], acc[m][n]);

        if (pre) store_tiles(sA[buf ^ 1], sB[buf ^ 1], tid, sr);
        __syncthreads();
    }

    const int hi = lane >> 4, ln = lane & 15;
#pragma unroll
    for (int m = 0; m < 4; ++m)
#pragma unroll
        for (int n = 0; n < 2; ++n)
#pragma unroll
            for (int r = 0; r < 8; ++r) {
                int row = bM + wm * 64 + m * 16 + r + 8 * hi;
                int col = bN + wn * 32 + n * 16 + ln;
                float v = acc[m][n][r] + bias[col];
                size_t idx = (size_t)row * N + col;
                if (EPI == 1) {
                    Cf[idx] = v + res[idx];
                } else if (EPI == 2) {
                    float g = 0.5f * v * (1.0f + erff(v * 0.7071067811865475f));
                    Ch[idx] = (_Float16)g;
                } else {
                    Ch[idx] = (_Float16)v;
                }
            }
}

// ---------------------------------------------------------------- attention
// Flash attention: 128 queries per block, key blocks of 64, HD=64.
#define ALD 72  // padded LDS row stride (144B)

struct KVRegs { v8h k0, k1, v0, v1; };

__device__ __forceinline__ void gload_kv(const _Float16* baseK, const _Float16* baseV,
                                         size_t rowstride, int kb, int tid, KVRegs& s) {
    int key = kb * 64 + (tid >> 2);
    int c = (tid & 3) * 8;
    s.k0 = *(const v8h*)(baseK + (size_t)key * rowstride + c);
    s.k1 = *(const v8h*)(baseK + (size_t)key * rowstride + c + 32);
    s.v0 = *(const v8h*)(baseV + (size_t)key * rowstride + c);
    s.v1 = *(const v8h*)(baseV + (size_t)key * rowstride + c + 32);
}

__device__ __forceinline__ void store_kv(_Float16* sK, _Float16* sVt, int tid, const KVRegs& s) {
    int key = tid >> 2, c = (tid & 3) * 8;
    *(v8h*)(sK + key * ALD + c) = s.k0;
    *(v8h*)(sK + key * ALD + c + 32) = s.k1;
#pragma unroll
    for (int j = 0; j < 8; ++j) sVt[(c + j) * ALD + key] = s.v0[j];
#pragma unroll
    for (int j = 0; j < 8; ++j) sVt[(c + 32 + j) * ALD + key] = s.v1[j];
}

__global__ __launch_bounds__(256, 1) void attn_flash(
        const _Float16* __restrict__ qkv, _Float16* __restrict__ ctx) {
    __shared__ __align__(16) _Float16 sQ[128 * ALD];
    __shared__ __align__(16) _Float16 sK[64 * ALD];
    __shared__ __align__(16) _Float16 sVt[64 * ALD];   // transposed: [d][key]
    __shared__ __align__(16) _Float16 sP[8 * 16 * ALD];

    const int tid = threadIdx.x, lane = tid & 31, wid = tid >> 5;
    const int qb = blockIdx.x * 128;
    const int head = blockIdx.y;
    const int bi = blockIdx.z;
    const size_t rowstride = 3 * H_DIM;
    const _Float16* baseQ = qkv + (size_t)bi * SEQ * rowstride + head * HD;
    const _Float16* baseK = baseQ + H_DIM;
    const _Float16* baseV = baseQ + 2 * H_DIM;

    // stage Q block [128][64], pre-scaled by 1/sqrt(HD)=0.125 (exact in f16)
    {
        const _Float16 qscale = (_Float16)0.125f;
        int r0 = tid >> 3, c = (tid & 7) * 8;
#pragma unroll
        for (int i = 0; i < 4; ++i) {
            int row = r0 + i * 32;
            v8h v = *(const v8h*)(baseQ + (size_t)(qb + row) * rowstride + c);
#pragma unroll
            for (int j = 0; j < 8; ++j) v[j] = v[j] * qscale;
            *(v8h*)(sQ + row * ALD + c) = v;
        }
    }

    KVRegs kv;
    gload_kv(baseK, baseV, rowstride, 0, tid, kv);
    __syncthreads();

    const v16h qf0 = ld_frag(sQ, wid * 16, ALD, 0);
    const v16h qf1 = ld_frag(sQ, wid * 16, ALD, 32);

    v8f oacc[4];
#pragma unroll
    for (int n = 0; n < 4; ++n) oacc[n] = vzero8();
    float mprev[8], lsum[8];
#pragma unroll
    for (int r = 0; r < 8; ++r) { mprev[r] = -1e30f; lsum[r] = 0.0f; }

    const int hi = lane >> 4, ln = lane & 15;
    _Float16* myP = sP + wid * 16 * ALD;

    for (int kb = 0; kb < SEQ / 64; ++kb) {
        store_kv(sK, sVt, tid, kv);
        __syncthreads();
        if (kb + 1 < SEQ / 64)
            gload_kv(baseK, baseV, rowstride, kb + 1, tid, kv);  // overlap with compute

        // scores (already scaled via Q): 16 q-rows x 64 keys per wave
        v8f sacc[4];
#pragma unroll
        for (int j = 0; j < 4; ++j) {
            sacc[j] = vzero8();
            v16h kf0 = ld_frag(sK, j * 16, ALD, 0);
            v16h kf1 = ld_frag(sK, j * 16, ALD, 32);
            sacc[j] = wmma16(qf0, kf0, sacc[j]);
            sacc[j] = wmma16(qf1, kf1, sacc[j]);
        }

        // online softmax; row stats align with accumulator layout
        float mnew[8];
#pragma unroll
        for (int r = 0; r < 8; ++r)
            mnew[r] = fmaxf(fmaxf(sacc[0][r], sacc[1][r]), fmaxf(sacc[2][r], sacc[3][r]));
        rowmax8(mnew);

        float alpha[8], rsum[8];
#pragma unroll
        for (int r = 0; r < 8; ++r) {
            float mt = fmaxf(mprev[r], mnew[r]);
            alpha[r] = __expf(mprev[r] - mt);
            mprev[r] = mt;
            rsum[r] = 0.0f;
        }
        // fused: exp -> LDS (A-layout staging) -> row-sum
#pragma unroll
        for (int j = 0; j < 4; ++j)
#pragma unroll
            for (int r = 0; r < 8; ++r) {
                float e = __expf(sacc[j][r] - mprev[r]);
                rsum[r] += e;
                myP[(r + 8 * hi) * ALD + j * 16 + ln] = (_Float16)e;
            }
        rowsum8(rsum);
#pragma unroll
        for (int r = 0; r < 8; ++r) lsum[r] = lsum[r] * alpha[r] + rsum[r];
#pragma unroll
        for (int n = 0; n < 4; ++n)
#pragma unroll
            for (int r = 0; r < 8; ++r) oacc[n][r] *= alpha[r];

        v16h pf0 = ld_frag(myP, 0, ALD, 0);
        v16h pf1 = ld_frag(myP, 0, ALD, 32);
#pragma unroll
        for (int n = 0; n < 4; ++n) {
            v16h vf0 = ld_frag(sVt, n * 16, ALD, 0);
            v16h vf1 = ld_frag(sVt, n * 16, ALD, 32);
            oacc[n] = wmma16(pf0, vf0, oacc[n]);
            oacc[n] = wmma16(pf1, vf1, oacc[n]);
        }
        __syncthreads();
    }

    // normalize (reciprocal per row) + write ctx [B,S,H] as f16
    float rinv[8];
#pragma unroll
    for (int r = 0; r < 8; ++r) rinv[r] = 1.0f / lsum[r];
#pragma unroll
    for (int n = 0; n < 4; ++n)
#pragma unroll
        for (int r = 0; r < 8; ++r) {
            int q = qb + wid * 16 + r + 8 * hi;
            float ov = oacc[n][r] * rinv[r];
            ctx[((size_t)(bi * SEQ + q)) * H_DIM + head * HD + n * 16 + ln] = (_Float16)ov;
        }
}

// ---------------------------------------------------------------- layernorm
template <bool WRITE_H>
__global__ __launch_bounds__(256) void layer_norm(
        const float* __restrict__ in, const float* __restrict__ g,
        const float* __restrict__ b, float* __restrict__ out,
        _Float16* __restrict__ outh) {
    __shared__ float r1[256], r2[256];
    const int row = blockIdx.x, tid = threadIdx.x;
    const float* rp = in + (size_t)row * H_DIM;
    float v[3], s = 0.0f, s2 = 0.0f;
#pragma unroll
    for (int i = 0; i < 3; ++i) {
        v[i] = rp[tid + i * 256];
        s += v[i];
        s2 += v[i] * v[i];
    }
    r1[tid] = s; r2[tid] = s2;
    __syncthreads();
    for (int off = 128; off > 0; off >>= 1) {
        if (tid < off) { r1[tid] += r1[tid + off]; r2[tid] += r2[tid + off]; }
        __syncthreads();
    }
    float mu = r1[0] * (1.0f / H_DIM);
    float var = r2[0] * (1.0f / H_DIM) - mu * mu;
    float rs = rsqrtf(var + LN_EPS);
#pragma unroll
    for (int i = 0; i < 3; ++i) {
        int c = tid + i * 256;
        float y = (v[i] - mu) * rs * g[c] + b[c];
        out[(size_t)row * H_DIM + c] = y;
        if (WRITE_H) outh[(size_t)row * H_DIM + c] = (_Float16)y;
    }
}

// ---------------------------------------------------------------- launch
extern "C" void kernel_launch(void* const* d_in, const int* in_sizes, int n_in,
                              void* d_out, int out_size, void* d_ws, size_t ws_size,
                              hipStream_t stream) {
    (void)in_sizes; (void)n_in; (void)out_size; (void)ws_size;
    const float* x     = (const float*)d_in[0];
    const float* w_qkv = (const float*)d_in[1];
    const float* b_qkv = (const float*)d_in[2];
    const float* w_out = (const float*)d_in[3];
    const float* b_out = (const float*)d_in[4];
    const float* w_ff1 = (const float*)d_in[5];
    const float* b_ff1 = (const float*)d_in[6];
    const float* w_ff2 = (const float*)d_in[7];
    const float* b_ff2 = (const float*)d_in[8];
    const float* g1    = (const float*)d_in[9];
    const float* be1   = (const float*)d_in[10];
    const float* g2    = (const float*)d_in[11];
    const float* be2   = (const float*)d_in[12];
    float* out = (float*)d_out;

    char* ws = (char*)d_ws;
    size_t off = 0;
    auto alloc = [&](size_t bytes) -> void* {
        void* p = ws + off;
        off = (off + bytes + 255) & ~(size_t)255;
        return p;
    };

    _Float16* xb    = (_Float16*)alloc((size_t)MTOT * H_DIM * 2);
    _Float16* wqkvh = (_Float16*)alloc((size_t)H_DIM * 3 * H_DIM * 2);
    _Float16* wouth = (_Float16*)alloc((size_t)H_DIM * H_DIM * 2);
    _Float16* wff1h = (_Float16*)alloc((size_t)H_DIM * FF_DIM * 2);
    _Float16* wff2h = (_Float16*)alloc((size_t)FF_DIM * H_DIM * 2);
    _Float16* qkvh  = (_Float16*)alloc((size_t)MTOT * 3 * H_DIM * 2);
    _Float16* ctxh  = (_Float16*)alloc((size_t)MTOT * H_DIM * 2);
    float*    tmp   = (float*)alloc((size_t)MTOT * H_DIM * 4);
    float*    x1    = (float*)alloc((size_t)MTOT * H_DIM * 4);
    _Float16* x1h   = (_Float16*)alloc((size_t)MTOT * H_DIM * 2);
    _Float16* hh    = (_Float16*)alloc((size_t)MTOT * FF_DIM * 2);

    auto cvt = [&](const float* src, _Float16* dst, size_t n) {
        int blocks = (int)((n + 1023) / 1024);
        cvt_f32_f16<<<blocks, 256, 0, stream>>>(src, dst, n);
    };
    cvt(x,     xb,    (size_t)MTOT * H_DIM);
    cvt(w_qkv, wqkvh, (size_t)H_DIM * 3 * H_DIM);
    cvt(w_out, wouth, (size_t)H_DIM * H_DIM);
    cvt(w_ff1, wff1h, (size_t)H_DIM * FF_DIM);
    cvt(w_ff2, wff2h, (size_t)FF_DIM * H_DIM);

    // QKV projection: [8192,768] @ [768,2304] + bias -> f16
    gemm_f16<0><<<dim3(3 * H_DIM / 128, MTOT / 128), 256, 0, stream>>>(
        xb, wqkvh, b_qkv, nullptr, nullptr, qkvh, MTOT, 3 * H_DIM, H_DIM);

    // flash attention -> ctx f16
    attn_flash<<<dim3(SEQ / 128, NHEAD, BATCH), 256, 0, stream>>>(qkvh, ctxh);

    // out projection + residual(x) -> tmp f32
    gemm_f16<1><<<dim3(H_DIM / 128, MTOT / 128), 256, 0, stream>>>(
        ctxh, wouth, b_out, x, tmp, nullptr, MTOT, H_DIM, H_DIM);

    // LN1 -> x1 (f32) + x1h (f16)
    layer_norm<true><<<MTOT, 256, 0, stream>>>(tmp, g1, be1, x1, x1h);

    // FF1 + exact GELU -> hh f16
    gemm_f16<2><<<dim3(FF_DIM / 128, MTOT / 128), 256, 0, stream>>>(
        x1h, wff1h, b_ff1, nullptr, nullptr, hh, MTOT, FF_DIM, H_DIM);

    // FF2 + residual(x1) -> tmp f32
    gemm_f16<1><<<dim3(H_DIM / 128, MTOT / 128), 256, 0, stream>>>(
        hh, wff2h, b_ff2, x1, tmp, nullptr, MTOT, H_DIM, FF_DIM);

    // LN2 -> final output f32
    layer_norm<false><<<MTOT, 256, 0, stream>>>(tmp, g2, be2, out, nullptr);
}
